// GraphNeuralNetworkGCP_23235773071704
// MI455X (gfx1250) — matrix operand order repaired and checked by reading, not compiled, not run on hardware
//
#include <hip/hip_runtime.h>
#include <hip/hip_bf16.h>

// ---------------------------------------------------------------------------
// GraphNeuralNetworkGCP on MI455X (gfx1250): bf16 WMMA GEMMs + streaming
// matvec, 32-step recurrence as a kernel chain.  B=8, V=2048, C=32, T=32.
// Memory-bound (AI ~3 flop/B): all big operands live as bf16 (~200MB/step,
// ~L2-resident vs 192MB L2 / 23.3 TB/s HBM).  Activations are zero-padded
// bf16 [16,K] tiles so the WMMA A-operand loads are branch-free VMEM (no
// EXEC churn, no WMMA->VALU WAR hazard NOPs in the inner loop).
// ---------------------------------------------------------------------------

#define GB 8        // batch
#define GV 2048     // vertices
#define GC 32       // colors
#define GT 32       // timesteps
#define GG (4 * GV) // gate width 8192

typedef __attribute__((ext_vector_type(16))) __bf16 v16bf;
typedef __attribute__((ext_vector_type(8)))  __bf16 v8bf;
typedef __attribute__((ext_vector_type(8)))  float  v8f;

__device__ __forceinline__ float sigm(float x) { return 1.0f / (1.0f + expf(-x)); }

// -------- A-operand: 16x32 bf16 tile from zero-padded bf16 acts [16,K] -----
// CDNA5 ISA 7.12.2 (16-bit A 16x32): lane l holds row m=l&15, half h=l>>4;
// elements e=0..15 -> K = k0+16h+e (contiguous order chosen here, not the ISA table order; matches load_B)
// Branch-free: padding rows 8..15 of X hold zeros.
__device__ __forceinline__ v16bf load_A(const __bf16* __restrict__ X, int K, int k0, int lane) {
  int m = lane & 15, h = lane >> 4;
  const __bf16* p = X + (size_t)m * K + k0 + 16 * h;
  v8bf lo = *(const v8bf*)(p);        // 16B aligned
  v8bf hi = *(const v8bf*)(p + 8);    // 16B aligned
  v16bf a;
#pragma unroll
  for (int e = 0; e < 8; ++e) { a[e] = lo[e]; a[8 + e] = hi[e]; }
  return a;
}

// -------- B-operand: 32x16 bf16 tile = W^T, W row-major [N,K] ---------------
// lane l holds column n=n0+(l&15), half h=l>>4, elements e -> K = k0+16h+e:
// 16 contiguous bf16 of row n of W -> one aligned 32-byte load per lane.
__device__ __forceinline__ v16bf load_B(const __bf16* __restrict__ W, int K, int n0, int k0, int lane) {
  int n = n0 + (lane & 15), h = lane >> 4;
  return *(const v16bf*)(W + (size_t)n * K + k0 + 16 * h);
}

// ---------------------------------------------------------------------------
// out[b,n] = act( X1@W1^T (+ X2@W2^T) + bias[n] (+ sc[b]*rowsum[n]) )
// X*: bf16 [16,K] zero-padded; W*: bf16 [N,K]; one wave per 16-wide N tile.
// act: 0=none, 1=relu, 2=sigmoid.  Two independent accumulators in the DUAL
// case so the two WMMAs per iteration have no D->C serial dependence.
// ---------------------------------------------------------------------------
template <bool DUAL>
__global__ void gemm_tile(const __bf16* __restrict__ X1, const __bf16* __restrict__ W1,
                          const __bf16* __restrict__ X2, const __bf16* __restrict__ W2,
                          const float* __restrict__ bias,
                          const float* __restrict__ sc, const float* __restrict__ rowsum,
                          float* __restrict__ out, int K, int N, int act) {
  int lane = threadIdx.x & 31;
  int wave = (int)((blockIdx.x * blockDim.x + threadIdx.x) >> 5);
  int n0 = wave * 16;
  v8f acc1 = {};
  v8f acc2 = {};
  for (int k0 = 0; k0 < K; k0 += 32) {
    __builtin_prefetch((const void*)(W1 + (size_t)(n0 + (lane & 15)) * K + k0 + 512), 0, 3);
    v16bf a1 = load_A(X1, K, k0, lane);
    v16bf b1 = load_B(W1, K, n0, k0, lane);
    acc1 = __builtin_amdgcn_wmma_f32_16x16x32_bf16(false, a1, false, b1, (short)0, acc1, false, false);
    if (DUAL) {
      v16bf a2 = load_A(X2, K, k0, lane);
      v16bf b2 = load_B(W2, K, n0, k0, lane);
      acc2 = __builtin_amdgcn_wmma_f32_16x16x32_bf16(false, a2, false, b2, (short)0, acc2, false, false);
    }
  }
  // C/D layout: VGPR r, lanes 0..15 hold M=r (real batch rows), N=n0+lane.
  if (lane < 16) {
    int n = n0 + lane;
    float bb = bias ? bias[n] : 0.0f;
    float rs = (sc != nullptr) ? rowsum[n] : 0.0f;
#pragma unroll
    for (int r = 0; r < 8; ++r) {
      float v = acc1[r] + bb + ((sc != nullptr) ? sc[r] * rs : 0.0f);
      if (DUAL) v += acc2[r];
      if (act == 1) v = fmaxf(v, 0.0f);
      else if (act == 2) v = sigm(v);
      out[(size_t)r * N + n] = v;
    }
  }
}

// ---- batched adjacency matvec: y[b,u] = Mvv[b,u,:] . x[b,:]  (all bf16) ----
__global__ void matvec_bf16(const __bf16* __restrict__ M, const __bf16* __restrict__ x,
                            __bf16* __restrict__ y) {
  int gw = (int)((blockIdx.x * blockDim.x + threadIdx.x) >> 5); // = b*GV + u
  int lane = threadIdx.x & 31;
  int b = gw >> 11;
  const __bf16* row = M + (size_t)gw * GV;
  const __bf16* xv  = x + (size_t)b * GV;
  float acc = 0.0f;
  for (int i = lane * 8; i < GV; i += 256) {
    v8bf m8 = *(const v8bf*)(row + i);
    v8bf x8 = *(const v8bf*)(xv + i);
#pragma unroll
    for (int e = 0; e < 8; ++e) acc += (float)m8[e] * (float)x8[e];
  }
#pragma unroll
  for (int off = 16; off; off >>= 1) acc += __shfl_xor(acc, off, 32);
  if (lane == 0) y[gw] = (__bf16)acc;   // rows 0..7 of padded [16,GV]
}

// ---- LSTM elementwise gate update (i,f,g,o); h stored bf16 for the GEMMs ---
__global__ void lstm_update(const float* __restrict__ gates, __bf16* __restrict__ h,
                            float* __restrict__ m) {
  int idx = blockIdx.x * blockDim.x + threadIdx.x; // [0, B*V)
  if (idx >= GB * GV) return;
  int b = idx / GV, v = idx % GV;
  const float* g = gates + (size_t)b * GG;
  float gi = sigm(g[v]);
  float gf = sigm(g[GV + v]);
  float gg = tanhf(g[2 * GV + v]);
  float go = sigm(g[3 * GV + v]);
  float cn = gf * m[idx] + gi * gg;
  m[idx] = cn;
  h[idx] = (__bf16)(go * tanhf(cn));
}

// ---- color stage A: s_c[b] = sum_c mask * relu(ch @ cmsg_w^T + b) ----------
__global__ void color_a(const float* __restrict__ ch, const float* __restrict__ cmsg_w,
                        const float* __restrict__ cmsg_b, const int* __restrict__ n_colors,
                        float* __restrict__ s_c) {
  int t = threadIdx.x;        // 256 threads: wave b handles batch b (wave32)
  int b = t >> 5, c = t & 31;
  float acc = cmsg_b[c];
#pragma unroll
  for (int j = 0; j < GC; ++j) acc += ch[b * GC + j] * cmsg_w[c * GC + j];
  acc = fmaxf(acc, 0.0f);
  float v = (c < n_colors[b]) ? acc : 0.0f;
#pragma unroll
  for (int off = 16; off; off >>= 1) v += __shfl_xor(v, off, 32);
  if (c == 0) s_c[b] = v;
}

// ---- color stage B: color LSTM cell (ping-pong ch/cm buffers) --------------
__global__ void color_b(const float* __restrict__ vsum, const int* __restrict__ n_colors,
                        const float* __restrict__ ch_in, const float* __restrict__ cm_in,
                        const float* __restrict__ c_wih, const float* __restrict__ c_whh,
                        const float* __restrict__ c_bih, const float* __restrict__ c_bhh,
                        float* __restrict__ ch_out, float* __restrict__ cm_out) {
  int t = threadIdx.x;
  int b = t >> 5, c = t & 31;
  int nc = n_colors[b];
  float vs = vsum[b];
  float g[4];
#pragma unroll
  for (int q = 0; q < 4; ++q) {
    int row = q * GC + c;
    float acc = c_bih[row] + c_bhh[row];
    for (int j = 0; j < GC; ++j) {
      float rin = (j < nc) ? vs : 0.0f;
      acc += rin * c_wih[row * GC + j] + ch_in[b * GC + j] * c_whh[row * GC + j];
    }
    g[q] = acc;
  }
  float gi = sigm(g[0]), gf = sigm(g[1]), gg = tanhf(g[2]), go = sigm(g[3]);
  float cn = gf * cm_in[b * GC + c] + gi * gg;
  cm_out[b * GC + c] = cn;
  ch_out[b * GC + c] = go * tanhf(cn);
}

// ---- one-time prep: split v0_wih -> rank-1 rowsums + bf16 Wv + folded bias -
__global__ void prep_v0(const float* __restrict__ wih, const float* __restrict__ bih,
                        const float* __restrict__ bhh, __bf16* __restrict__ Wv,
                        float* __restrict__ rowsum, float* __restrict__ bias0) {
  int wave = (int)((blockIdx.x * blockDim.x + threadIdx.x) >> 5); // one wave per row of 8192
  int lane = threadIdx.x & 31;
  const float* row = wih + (size_t)wave * (2 * GV);
  float s = 0.0f;
  for (int k = lane; k < GV; k += 32) {
    s += row[k];                                     // color-broadcast block
    Wv[(size_t)wave * GV + k] = (__bf16)row[GV + k]; // vertex block -> bf16
  }
#pragma unroll
  for (int off = 16; off; off >>= 1) s += __shfl_xor(s, off, 32);
  if (lane == 0) { rowsum[wave] = s; bias0[wave] = bih[wave] + bhh[wave]; }
}

__global__ void cvt_bf16(const float* __restrict__ src, __bf16* __restrict__ dst, size_t n) {
  size_t i = (size_t)blockIdx.x * blockDim.x + threadIdx.x;
  size_t stride = (size_t)gridDim.x * blockDim.x;
  for (; i < n; i += stride) dst[i] = (__bf16)src[i];
}

__global__ void add2(const float* __restrict__ a, const float* __restrict__ b,
                     float* __restrict__ o, int n) {
  int i = blockIdx.x * blockDim.x + threadIdx.x;
  if (i < n) o[i] = a[i] + b[i];
}

// mode 0: out[b] = sum_i x[b,i];  mode 1: out[b] = sigmoid(mean_i x[b,i])
__global__ void reduce_row(const float* __restrict__ x, float* __restrict__ out, int n, int mode) {
  int b = blockIdx.x, tid = threadIdx.x;
  float s = 0.0f;
  for (int i = tid; i < n; i += blockDim.x) s += x[(size_t)b * n + i];
#pragma unroll
  for (int off = 16; off; off >>= 1) s += __shfl_xor(s, off, 32);
  __shared__ float red[32];
  if ((tid & 31) == 0) red[tid >> 5] = s;
  __syncthreads();
  if (tid == 0) {
    float t = 0.0f;
    int nw = blockDim.x >> 5;
    for (int j = 0; j < nw; ++j) t += red[j];
    if (mode == 1) t = sigm(t / (float)n);
    out[b] = t;
  }
}

// ---------------------------------------------------------------------------
extern "C" void kernel_launch(void* const* d_in, const int* in_sizes, int n_in,
                              void* d_out, int out_size, void* d_ws, size_t ws_size,
                              hipStream_t stream) {
  (void)in_sizes; (void)n_in; (void)out_size; (void)ws_size;
  const float* Mvv      = (const float*)d_in[0];
  const int*   n_colors = (const int*)d_in[1];
  const float* vh0      = (const float*)d_in[2];
  const float* ch0      = (const float*)d_in[3];
  const float* v0_wih   = (const float*)d_in[4];
  const float* v0_whh   = (const float*)d_in[5];
  const float* v0_bih   = (const float*)d_in[6];
  const float* v0_bhh   = (const float*)d_in[7];
  const float* v1_wih   = (const float*)d_in[8];
  const float* v1_whh   = (const float*)d_in[9];
  const float* v1_bih   = (const float*)d_in[10];
  const float* v1_bhh   = (const float*)d_in[11];
  const float* c_wih    = (const float*)d_in[12];
  const float* c_whh    = (const float*)d_in[13];
  const float* c_bih    = (const float*)d_in[14];
  const float* c_bhh    = (const float*)d_in[15];
  const float* cmsg_w   = (const float*)d_in[16];
  const float* cmsg_b   = (const float*)d_in[17];
  const float* vmsg_w   = (const float*)d_in[18];
  const float* vmsg_b   = (const float*)d_in[19];
  const float* vote_w   = (const float*)d_in[20];
  const float* vote_b   = (const float*)d_in[21];

  // -------- workspace carve-up (256B aligned) --------
  char* w = (char*)d_ws;
  auto carve = [&](size_t bytes) { char* p = w; w += (bytes + 255) & ~(size_t)255; return p; };
  __bf16* Wv0   = (__bf16*)carve((size_t)GG * GV * 2); // v0_wih vertex half
  __bf16* Whh0  = (__bf16*)carve((size_t)GG * GV * 2);
  __bf16* Wih1  = (__bf16*)carve((size_t)GG * GV * 2);
  __bf16* Whh1  = (__bf16*)carve((size_t)GG * GV * 2);
  __bf16* Wvmsg = (__bf16*)carve((size_t)GV * GV * 2);
  __bf16* Wvote = (__bf16*)carve((size_t)GV * GV * 2);
  __bf16* MvvBf = (__bf16*)carve((size_t)GB * GV * GV * 2);
  float* rowsum_c = (float*)carve(GG * 4);
  float* bias0    = (float*)carve(GG * 4);
  float* bias1    = (float*)carve(GG * 4);
  // zero-padded [16, GV] bf16 activation tiles (rows 8..15 stay zero)
  __bf16* h0b  = (__bf16*)carve((size_t)16 * GV * 2);
  __bf16* h1b  = (__bf16*)carve((size_t)16 * GV * 2);
  __bf16* mvb  = (__bf16*)carve((size_t)16 * GV * 2);  // muled_v
  float* m0    = (float*)carve((size_t)GB * GV * 4);
  float* m1    = (float*)carve((size_t)GB * GV * 4);
  float* chP[2] = {(float*)carve(GB * GC * 4), (float*)carve(GB * GC * 4)};
  float* cmP[2] = {(float*)carve(GB * GC * 4), (float*)carve(GB * GC * 4)};
  float* s_c     = (float*)carve(GB * 4);
  float* vsum    = (float*)carve(GB * 4);
  float* vmbuf   = (float*)carve((size_t)GB * GV * 4);
  float* gates0  = (float*)carve((size_t)GB * GG * 4);
  float* gates1  = (float*)carve((size_t)GB * GG * 4);
  float* votebuf = (float*)carve((size_t)GB * GV * 4);

  // -------- one-time prep: bf16 conversions + rank-1 split + bias folding ----
  prep_v0<<<1024, 256, 0, stream>>>(v0_wih, v0_bih, v0_bhh, Wv0, rowsum_c, bias0);
  add2<<<(GG + 255) / 256, 256, 0, stream>>>(v1_bih, v1_bhh, bias1, GG);
  const size_t nW = (size_t)GG * GV;      // 16.7M
  const size_t nS = (size_t)GV * GV;      // 4.2M
  const size_t nM = (size_t)GB * GV * GV; // 33.5M
  cvt_bf16<<<8192, 256, 0, stream>>>(v0_whh, Whh0, nW);
  cvt_bf16<<<8192, 256, 0, stream>>>(v1_wih, Wih1, nW);
  cvt_bf16<<<8192, 256, 0, stream>>>(v1_whh, Whh1, nW);
  cvt_bf16<<<4096, 256, 0, stream>>>(vmsg_w, Wvmsg, nS);
  cvt_bf16<<<4096, 256, 0, stream>>>(vote_w, Wvote, nS);
  cvt_bf16<<<16384, 256, 0, stream>>>(Mvv, MvvBf, nM);

  // -------- state init: zero padded tiles, then fill rows 0..7 --------
  hipMemsetAsync(h0b, 0, (size_t)16 * GV * 2, stream);
  hipMemsetAsync(h1b, 0, (size_t)16 * GV * 2, stream);
  hipMemsetAsync(mvb, 0, (size_t)16 * GV * 2, stream);
  cvt_bf16<<<64, 256, 0, stream>>>(vh0, h0b, (size_t)GB * GV);
  cvt_bf16<<<64, 256, 0, stream>>>(vh0 + (size_t)GB * GV, h1b, (size_t)GB * GV);
  hipMemcpyAsync(chP[0], ch0, (size_t)GB * GC * 4, hipMemcpyDeviceToDevice, stream);
  hipMemsetAsync(m0, 0, (size_t)GB * GV * 4, stream);
  hipMemsetAsync(m1, 0, (size_t)GB * GV * 4, stream);
  hipMemsetAsync(cmP[0], 0, (size_t)GB * GC * 4, stream);

  // -------- recurrence --------
  int cur = 0;
  for (int t = 0; t < GT; ++t) {
    // s_c from old ch
    color_a<<<1, 256, 0, stream>>>(chP[cur], cmsg_w, cmsg_b, n_colors, s_c);
    // muled_v = Mvv @ h1   (8 * 2048 waves, one per adjacency row)
    matvec_bf16<<<(GB * GV) / 8, 256, 0, stream>>>(MvvBf, h1b, mvb);
    // vertex_msg = relu(h1 @ vmsg_w^T + b), then row-sums -> vsum
    gemm_tile<false><<<GV / 64, 128, 0, stream>>>(h1b, Wvmsg, nullptr, nullptr, vmsg_b,
                                                  nullptr, nullptr, vmbuf, GV, GV, 1);
    reduce_row<<<GB, 256, 0, stream>>>(vmbuf, vsum, GV, 0);
    // cell 0 gates: muled_v@Wv^T + h0@Whh0^T + s_c*rowsum_c + bias0
    gemm_tile<true><<<GG / 64, 128, 0, stream>>>(mvb, Wv0, h0b, Whh0, bias0,
                                                 s_c, rowsum_c, gates0, GV, GG, 0);
    lstm_update<<<(GB * GV) / 256, 256, 0, stream>>>(gates0, h0b, m0);
    // cell 1 gates: h0_new@Wih1^T + h1_old@Whh1^T + bias1
    gemm_tile<true><<<GG / 64, 128, 0, stream>>>(h0b, Wih1, h1b, Whh1, bias1,
                                                 nullptr, nullptr, gates1, GV, GG, 0);
    // color LSTM (uses old ch/cm, writes ping-pong partner)
    color_b<<<1, 256, 0, stream>>>(vsum, n_colors, chP[cur], cmP[cur],
                                   c_wih, c_whh, c_bih, c_bhh, chP[1 - cur], cmP[1 - cur]);
    // h1/m1 update last (all h1 consumers above already ran)
    lstm_update<<<(GB * GV) / 256, 256, 0, stream>>>(gates1, h1b, m1);
    cur ^= 1;
  }

  // -------- readout: pred = sigmoid(mean(sigmoid(h1 @ vote_w^T + b))) --------
  gemm_tile<false><<<GV / 64, 128, 0, stream>>>(h1b, Wvote, nullptr, nullptr, vote_b,
                                                nullptr, nullptr, votebuf, GV, GV, 2);
  reduce_row<<<GB, 256, 0, stream>>>(votebuf, (float*)d_out, GV, 1);
}
